// GConvMultiHeadAttention_81303730913417
// MI455X (gfx1250) — matrix operand order repaired and checked by reading, not compiled, not run on hardware
//
#include <hip/hip_runtime.h>

// ---------------------------------------------------------------------------
// GConv Multi-Head Attention for MI455X (gfx1250, wave32, WMMA bf16,
// async global->LDS staging with double buffering)
// B=8, C=512, L=1024, H=8, D=64
// ---------------------------------------------------------------------------

#define Bsz 8
#define Cch 512
#define Lseq 1024
#define Hh 8
#define Dd 64

typedef __bf16 v16bf __attribute__((ext_vector_type(16)));
typedef float  v8f   __attribute__((ext_vector_type(8)));

// RNE float -> bf16
__device__ inline unsigned short f2bf(float f) {
  unsigned u = __float_as_uint(f);
  u += 0x7FFFu + ((u >> 16) & 1u);
  return (unsigned short)(u >> 16);
}

// ---------------------------------------------------------------------------
// CDNA5 async global->LDS copy (ASYNCcnt-tracked), per-lane 16B.
// Generic LDS pointer low 32 bits == LDS byte address (ISA 10.2 aperture map).
// ---------------------------------------------------------------------------
__device__ inline void async_ld_b128(unsigned lds_addr, const void* base,
                                     int voff) {
  asm volatile("global_load_async_to_lds_b128 %0, %1, %2"
               :: "v"(lds_addr), "v"(voff), "s"(base) : "memory");
}
__device__ inline void wait_async0() {
  asm volatile("s_wait_asynccnt 0" ::: "memory");
}

// ---------------------------------------------------------------------------
// Kernel 1: grouped conv1d (k=3, reflect pad, no bias) -> bf16.
// transposed=1: out is [b][h][l][d] (feeds async WMMA staging for Q/K)
// transposed=0: out is [b][c][l]    (V: rows already contiguous along l)
// grid: (L/128, B*H), block 256.
// ---------------------------------------------------------------------------
__global__ __launch_bounds__(256)
void gconv_kernel(const float* __restrict__ x, const float* __restrict__ w,
                  unsigned short* __restrict__ out, float scale,
                  int transposed) {
  __shared__ float xs[64 * 130];     // [ci][l0-1 .. l0+128]
  __shared__ float wl[64 * 64 * 3];  // [co_local][ci][k]

  const int tid = threadIdx.x;
  const int b  = blockIdx.y >> 3;
  const int g  = blockIdx.y & 7;
  const int l0 = blockIdx.x << 7;

  for (int idx = tid; idx < 64 * 130; idx += 256) {
    int ci = idx / 130;
    int j  = idx - ci * 130;
    int gl = l0 + j - 1;
    if (gl < 0) gl = -gl;                    // reflect: -1 -> 1
    if (gl >= Lseq) gl = 2 * Lseq - 2 - gl;  // reflect: L -> L-2
    xs[idx] = x[((b * Cch + g * Dd + ci) << 10) + gl];
  }
  for (int idx = tid; idx < 64 * 192; idx += 256)
    wl[idx] = w[g * (64 * 192) + idx];
  __syncthreads();

  const int l   = tid & 127;
  const int coh = tid >> 7;

  float acc[32];
#pragma unroll
  for (int i = 0; i < 32; ++i) acc[i] = 0.f;

  for (int ci = 0; ci < 64; ++ci) {
    float x0 = xs[ci * 130 + l];
    float x1 = xs[ci * 130 + l + 1];
    float x2 = xs[ci * 130 + l + 2];
    const float* wr = &wl[(coh * 32) * 192 + ci * 3];
#pragma unroll
    for (int co = 0; co < 32; ++co) {
      acc[co] += wr[co * 192 + 0] * x0 + wr[co * 192 + 1] * x1 +
                 wr[co * 192 + 2] * x2;
    }
  }

  if (transposed) {
    // [b*H+g][l][d], pack bf16 pairs -> 64B contiguous per thread
    int base = ((b * Hh + g) << 16) + ((l0 + l) << 6) + coh * 32;  // elems
    unsigned* o32 = (unsigned*)out;
#pragma unroll
    for (int i = 0; i < 16; ++i) {
      unsigned lo = f2bf(acc[2 * i] * scale);
      unsigned hi = f2bf(acc[2 * i + 1] * scale);
      o32[(base >> 1) + i] = lo | (hi << 16);
    }
  } else {
#pragma unroll
    for (int co = 0; co < 32; ++co) {
      int cg = g * Dd + coh * 32 + co;
      out[((b * Cch + cg) << 10) + l0 + l] = f2bf(acc[co] * scale);
    }
  }
}

// ---------------------------------------------------------------------------
// WMMA helpers (CDNA5 16x16x32 bf16, f32 accumulate)
// ---------------------------------------------------------------------------
__device__ inline v8f wmma_bf16(v16bf a, v16bf b, v8f c) {
  return __builtin_amdgcn_wmma_f32_16x16x32_bf16(false, a, false, b,
                                                 (short)0, c, false, false);
}

// A-fragment from 64-elem (32-dword) swizzled LDS rows.
// A layout used here (contiguous k, same as B): lane->M, VGPR i + lane-half -> K pairs:
// kb(i,half) = 2i + 16*half
// Row chunk swizzle: 16B chunk j>>2 xor (row&7).
__device__ inline v16bf load_a64(const unsigned* p, int row, int kofs,
                                 int half) {
  union { v16bf v; unsigned u[8]; } f;
  int rb = row << 5;
  int sw = row & 7;
#pragma unroll
  for (int i = 0; i < 8; ++i) {
    int kb = 2 * i + 16 * half + kofs;
    int j  = kb >> 1;
    f.u[i] = p[rb + ((((j >> 2) ^ sw) << 2) | (j & 3))];
  }
  return f.v;
}

// B-fragment: 8 dwords (two 16B chunks) of one swizzled row.
// B layout: lane -> N + K-half; VGPR i -> K=2i,2i+1 within the half.
__device__ inline v16bf load_b_row(const unsigned* p, int rowbase_dw,
                                   int jbase, int sw) {
  union { v16bf v; unsigned u[8]; } f;
  int c0 = (((jbase >> 2) ^ sw) << 2);
  int c1 = ((((jbase >> 2) + 1) ^ sw) << 2);
#pragma unroll
  for (int i = 0; i < 4; ++i) f.u[i] = p[rowbase_dw + c0 + i];
#pragma unroll
  for (int i = 0; i < 4; ++i) f.u[4 + i] = p[rowbase_dw + c1 + i];
  return f.v;
}

// A-fragment from linear-pitch LDS (P tiles, pitch in elements, even)
__device__ inline v16bf load_a_lin(const unsigned* p, int row_elem_base,
                                   int half) {
  union { v16bf v; unsigned u[8]; } f;
#pragma unroll
  for (int i = 0; i < 8; ++i) {
    int kb = 2 * i + 16 * half;
    f.u[i] = p[(row_elem_base + kb) >> 1];
  }
  return f.v;
}

// ---------------------------------------------------------------------------
// Kernel 2: causal flash attention per (b, h, 128-row query block).
// 8 waves x 16 rows. Async-staged, double-buffered K/V chunks.
// Writes O (pre-residual) f32 [b][l][c].
// ---------------------------------------------------------------------------
#define PP 34  // P pitch (elements): 17-dword step, conflict-free

__global__ __launch_bounds__(256)
void attn_kernel(const unsigned short* __restrict__ qT,
                 const unsigned short* __restrict__ kT,
                 const unsigned short* __restrict__ vN,
                 float* __restrict__ Ow) {
  __shared__ unsigned short Qt[128 * 64];      // [l][d], chunk-swizzled
  __shared__ unsigned short Kt[2][128 * 64];   // [m][d], chunk-swizzled
  __shared__ unsigned short Vs[2][64 * 128];   // [d][m], chunk-swizzled
  __shared__ unsigned short Pb[8 * 16 * PP];   // per-wave P tiles

  const int tid  = threadIdx.x;
  const int b    = blockIdx.y >> 3;
  const int h    = blockIdx.y & 7;
  const int lb   = blockIdx.x;
  const int w    = __builtin_amdgcn_readfirstlane(tid >> 5);
  const int lane = tid & 31;
  const int n    = lane & 15;
  const int half = lane >> 4;

  const int tbase = (b * Hh + h) << 16;        // [b][h] base, transposed elems
  const int vbase = (b * Cch + h * Dd) << 10;  // [b][h*64] base, [c][l] elems

  // ---- async stage: Q block + chunk 0 of K/V into buffer 0 --------------
  for (int idx = tid; idx < 1024; idx += 256) {
    int row = idx >> 3, d8 = idx & 7;
    unsigned lds =
        (unsigned)(size_t)&Qt[(row << 6) + ((d8 ^ (row & 7)) << 3)];
    async_ld_b128(lds, qT, (tbase + ((lb * 128 + row) << 6) + (d8 << 3)) * 2);
  }
  for (int idx = tid; idx < 1024; idx += 256) {
    int row = idx >> 3, d8 = idx & 7;
    unsigned lds =
        (unsigned)(size_t)&Kt[0][(row << 6) + ((d8 ^ (row & 7)) << 3)];
    async_ld_b128(lds, kT, (tbase + (row << 6) + (d8 << 3)) * 2);
  }
  for (int idx = tid; idx < 1024; idx += 256) {
    int d = idx >> 4, c16 = idx & 15;
    unsigned lds =
        (unsigned)(size_t)&Vs[0][(d << 7) + ((c16 ^ (d & 15)) << 3)];
    async_ld_b128(lds, vN, (vbase + (d << 10) + (c16 << 3)) * 2);
  }
  wait_async0();
  __syncthreads();

  const unsigned* Qd = (const unsigned*)Qt;
  const unsigned* Pd = (const unsigned*)Pb;

  v16bf aq0 = load_a64(Qd, w * 16 + n, 0, half);
  v16bf aq1 = load_a64(Qd, w * 16 + n, 32, half);

  v8f o[4];
#pragma unroll
  for (int t = 0; t < 4; ++t) o[t] = (v8f){};
  float rm[8], rs[8];
#pragma unroll
  for (int r = 0; r < 8; ++r) { rm[r] = -1e38f; rs[r] = 0.f; }

  const int lmaxw = lb * 128 + w * 16 + 15;
  const int lgb   = lb * 128 + w * 16 + 8 * half;

  for (int ch = 0; ch <= lb; ++ch) {
    const int cur = ch & 1;
    if (ch < lb) {  // prefetch next chunk into the other buffer
      const int nb = cur ^ 1;
      for (int idx = tid; idx < 1024; idx += 256) {
        int row = idx >> 3, d8 = idx & 7;
        unsigned lds =
            (unsigned)(size_t)&Kt[nb][(row << 6) + ((d8 ^ (row & 7)) << 3)];
        async_ld_b128(lds, kT,
                      (tbase + (((ch + 1) * 128 + row) << 6) + (d8 << 3)) * 2);
      }
      for (int idx = tid; idx < 1024; idx += 256) {
        int d = idx >> 4, c16 = idx & 15;
        unsigned lds =
            (unsigned)(size_t)&Vs[nb][(d << 7) + ((c16 ^ (d & 15)) << 3)];
        async_ld_b128(lds, vN,
                      (vbase + (d << 10) + (ch + 1) * 128 + (c16 << 3)) * 2);
      }
    }

    const unsigned* Kd = (const unsigned*)&Kt[cur][0];
    const unsigned* Vd = (const unsigned*)&Vs[cur][0];

    for (int mt = 0; mt < 4; ++mt) {
      int mg0 = ch * 128 + mt * 32;
      if (mg0 > lmaxw) break;  // wave-uniform causal skip

      // ---- S = Q^T K, two 16-col subtiles ----------------------------
      v8f s0 = (v8f){}, s1 = (v8f){};
      {
        int r0 = mt * 32 + n, r1 = r0 + 16;
        int jb = half * 8;
        s0 = wmma_bf16(aq0, load_b_row(Kd, r0 << 5, jb, r0 & 7), s0);
        s0 = wmma_bf16(aq1, load_b_row(Kd, r0 << 5, 16 + jb, r0 & 7), s0);
        s1 = wmma_bf16(aq0, load_b_row(Kd, r1 << 5, jb, r1 & 7), s1);
        s1 = wmma_bf16(aq1, load_b_row(Kd, r1 << 5, 16 + jb, r1 & 7), s1);
      }

      // ---- causal mask + online softmax ------------------------------
#pragma unroll
      for (int r = 0; r < 8; ++r) {
        int lg = lgb + r;
        if (mg0 + n > lg)      s0[r] = -1e30f;
        if (mg0 + 16 + n > lg) s1[r] = -1e30f;

        float tm = fmaxf(s0[r], s1[r]);
#pragma unroll
        for (int off = 1; off < 16; off <<= 1)
          tm = fmaxf(tm, __shfl_xor(tm, off));
        float newm  = fmaxf(rm[r], tm);
        float scale = __expf(rm[r] - newm);
        rm[r] = newm;
        float p0 = __expf(s0[r] - newm);
        float p1 = __expf(s1[r] - newm);
        float ts = p0 + p1;
#pragma unroll
        for (int off = 1; off < 16; off <<= 1) ts += __shfl_xor(ts, off);
        rs[r] = rs[r] * scale + ts;
#pragma unroll
        for (int t = 0; t < 4; ++t) o[t][r] *= scale;

        int pbase = (w * 16 + r + 8 * half) * PP + n;
        Pb[pbase]      = f2bf(p0);
        Pb[pbase + 16] = f2bf(p1);
      }
      // LDS ops are in-order within a wave: safe to read P back now.
      v16bf ap = load_a_lin(Pd, (w * 16 + n) * PP, half);

      // ---- O += P V^T over 4 d-tiles ---------------------------------
#pragma unroll
      for (int t = 0; t < 4; ++t) {
        v16bf bv = load_b_row(Vd, (t * 16 + n) << 6, mt * 16 + half * 8,
                              (t * 16 + n) & 15);
        o[t] = wmma_bf16(ap, bv, o[t]);
      }
    }

    wait_async0();    // next chunk landed (and our issues retired)
    __syncthreads();  // all waves done with 'cur' before it is overwritten
  }

  // ---- normalize + store O as f32 [b][l][c] ---------------------------
#pragma unroll
  for (int r = 0; r < 8; ++r) {
    float inv = 1.0f / rs[r];
    int l = lgb + r;
#pragma unroll
    for (int t = 0; t < 4; ++t) {
      int c = h * Dd + t * 16 + n;
      Ow[(((b << 10) + l) << 9) + c] = o[t][r] * inv;
    }
  }
}

// ---------------------------------------------------------------------------
// Kernel 3: residual add + LayerNorm over C. grid: (L/32, B), block 256.
// ---------------------------------------------------------------------------
#define LNP 513

__global__ __launch_bounds__(256)
void ln_kernel(const float* __restrict__ qin, const float* __restrict__ Ow,
               const float* __restrict__ gamma, const float* __restrict__ beta,
               float* __restrict__ out) {
  __shared__ float xb[32 * LNP];
  const int tid = threadIdx.x;
  const int b   = blockIdx.y;
  const int l0  = blockIdx.x << 5;

  for (int idx = tid; idx < 32 * Cch; idx += 256) {
    int l = idx >> 9, c = idx & 511;
    xb[l * LNP + c] = Ow[(((b << 10) + l0 + l) << 9) + c];
  }
  __syncthreads();
  for (int idx = tid; idx < 32 * Cch; idx += 256) {
    int c = idx >> 5, l = idx & 31;
    xb[l * LNP + c] += qin[((b * Cch + c) << 10) + l0 + l];
  }
  __syncthreads();

  const int row  = tid >> 3;
  const int part = tid & 7;
  float s = 0.f, sq = 0.f;
  for (int i = 0; i < 64; ++i) {
    float x = xb[row * LNP + part * 64 + i];
    s += x; sq += x * x;
  }
#pragma unroll
  for (int off = 1; off < 8; off <<= 1) {
    s  += __shfl_xor(s, off);
    sq += __shfl_xor(sq, off);
  }
  float mu   = s * (1.0f / Cch);
  float var  = sq * (1.0f / Cch) - mu * mu;
  float rsig = rsqrtf(var + 1e-5f);

  for (int i = 0; i < 64; ++i) {
    int c = part * 64 + i;
    float x = xb[row * LNP + c];
    xb[row * LNP + c] = (x - mu) * rsig * gamma[c] + beta[c];
  }
  __syncthreads();

  for (int idx = tid; idx < 32 * Cch; idx += 256) {
    int c = idx >> 5, l = idx & 31;
    out[((b * Cch + c) << 10) + l0 + l] = xb[l * LNP + c];
  }
}

// ---------------------------------------------------------------------------
// launch
// ---------------------------------------------------------------------------
extern "C" void kernel_launch(void* const* d_in, const int* in_sizes, int n_in,
                              void* d_out, int out_size, void* d_ws,
                              size_t ws_size, hipStream_t stream) {
  const float* q     = (const float*)d_in[0];
  const float* k     = (const float*)d_in[1];
  // d_in[2] = mask (causal, applied analytically)
  const float* wq    = (const float*)d_in[3];
  const float* wk    = (const float*)d_in[4];
  const float* wv    = (const float*)d_in[5];
  const float* gamma = (const float*)d_in[6];
  const float* beta  = (const float*)d_in[7];
  float* out = (float*)d_out;

  // workspace: qT/kT bf16 [b][h][l][d], v bf16 [b][c][l], O f32 [b][l][c]
  char* ws = (char*)d_ws;
  unsigned short* qws = (unsigned short*)(ws);
  unsigned short* kws = (unsigned short*)(ws + (size_t)8 * 1024 * 1024);
  unsigned short* vws = (unsigned short*)(ws + (size_t)16 * 1024 * 1024);
  float*          Ow  = (float*)(ws + (size_t)24 * 1024 * 1024);

  dim3 blk(256);
  dim3 g1(Lseq / 128, Bsz * Hh);
  gconv_kernel<<<g1, blk, 0, stream>>>(q, wq, qws, 0.35355339059327373f, 1);
  gconv_kernel<<<g1, blk, 0, stream>>>(k, wk, kws, 1.0f, 1);
  gconv_kernel<<<g1, blk, 0, stream>>>(k, wv, vws, 1.0f, 0);  // v input == k

  dim3 g2(Lseq / 128, Bsz * Hh);
  attn_kernel<<<g2, blk, 0, stream>>>(qws, kws, vws, Ow);

  dim3 g3(Lseq / 32, Bsz);
  ln_kernel<<<g3, blk, 0, stream>>>(q, Ow, gamma, beta, out);
}
